// LOBSNNModel_80573586473177
// MI455X (gfx1250) — compile-verified, matrix operand order checked by reading, not run on hardware
//
#include <hip/hip_runtime.h>

#define B_   128
#define T_   1024
#define TP   (T_ + 2)      // time-padded rows (zero row at 0 and T+1)
#define CIN  64
#define CC   256
#define H_   1024
#define NCLS 3
#define NBLK 64            // persistent scan grid

#define THR    1.0f
#define BETA1  0.95f
#define ALPHA2 0.9f
#define BETA2  0.95f
#define EPSBN  1e-5f

typedef __attribute__((ext_vector_type(16))) _Float16 v16h;
typedef __attribute__((ext_vector_type(8)))  float    v8f;

union F16x16 { v16h v; uint4 q[2]; };

__device__ __forceinline__ v8f wmma_f16(const F16x16& a, const F16x16& b, v8f c) {
    return __builtin_amdgcn_wmma_f32_16x16x32_f16(false, a.v, false, b.v,
                                                  (short)0, c, false, false);
}

// ---------------------------------------------------------------------------
// Weight pre-swizzle into WMMA B-operand tile layout.
// B[k][n] = src[n*sn + k*sk + off].  Packed dword i:
//   v=i&7, lane=(i>>3)&31, tile=i>>8, nt=tile%NT, kt=tile/NT
//   n = nt*16 + (lane&15)
//   k = kt*32 + ((v&4)?16:0) + ((lane>=16)?8:0) + (v&3)*2   (pair k,k+1)
// ---------------------------------------------------------------------------
__global__ void pack_b_kernel(const float* __restrict__ src, _Float16* __restrict__ dst,
                              int KT, int NT, int sn, int sk, int off) {
    int i = blockIdx.x * blockDim.x + threadIdx.x;
    int total = KT * NT * 256;
    if (i >= total) return;
    int v = i & 7, lane = (i >> 3) & 31, tile = i >> 8;
    int nt = tile % NT, kt = tile / NT;
    int n = nt * 16 + (lane & 15);
    int k = kt * 32 + ((v & 4) ? 16 : 0) + ((lane >> 4) ? 8 : 0) + (v & 3) * 2;
    dst[2 * i]     = (_Float16)src[(size_t)n * sn + (size_t)k * sk + off];
    dst[2 * i + 1] = (_Float16)src[(size_t)n * sn + (size_t)(k + 1) * sk + off];
}

// x [B][T][CIN] f32 -> padded [B][TP][CIN] f16 (row 1..T)
__global__ void cvt_x_pad(const float* __restrict__ src, _Float16* __restrict__ dst) {
    size_t i = (size_t)blockIdx.x * blockDim.x + threadIdx.x;
    size_t per = (size_t)T_ * CIN;
    size_t b = i / per, r = i % per;
    dst[(b * TP + 1) * CIN + r] = (_Float16)src[i];
}

// ---------------------------------------------------------------------------
// Conv (k=3, pad=1) as 3 shifted GEMMs on time-padded input [B][TP][Cin] f16.
// One wave -> 16(t) x 16(co) output tile. Branch-free inner loop.
// ---------------------------------------------------------------------------
__global__ void conv_kernel(const _Float16* __restrict__ act, const _Float16* __restrict__ wp,
                            const float* __restrict__ bias,
                            float* __restrict__ of32, _Float16* __restrict__ of16,
                            int Cin, int doRelu) {
    int tid  = blockIdx.x * blockDim.x + threadIdx.x;
    int lane = threadIdx.x & 31;
    int w    = tid >> 5;
    int cot  = w & 15;          // CC/16 = 16
    int tt   = (w >> 4) & 63;   // T/16  = 64
    int b    = w >> 10;
    int m    = lane & 15, half = lane >> 4, kb = half * 8, n = m;
    int t0   = tt * 16;
    int KT   = Cin >> 5;
    v8f acc = {};
    for (int d = 0; d < 3; ++d) {
        const _Float16* arow = act + ((size_t)b * TP + t0 + m + d) * Cin;
        const _Float16* wpd  = wp + (size_t)d * ((size_t)KT * 16 * 512);
#pragma unroll
        for (int kt = 0; kt < KT; ++kt) {
            F16x16 a, bm;
            a.q[0] = *(const uint4*)(arow + kt * 32 + kb);
            a.q[1] = *(const uint4*)(arow + kt * 32 + kb + 16);
            const uint4* p = (const uint4*)wpd + (((size_t)(kt * 16 + cot) * 32 + lane) * 2);
            bm.q[0] = p[0]; bm.q[1] = p[1];
            acc = wmma_f16(a, bm, acc);
        }
    }
    int co0 = cot * 16;
    float bs = bias[co0 + n];
#pragma unroll
    for (int v = 0; v < 8; ++v) {
        int t = t0 + v + half * 8;
        float val = acc[v] + bs;
        if (doRelu) val = fmaxf(val, 0.f);
        size_t o = ((size_t)b * T_ + t) * CC + co0 + n;         // h1 unpadded
        if (of32) of32[o] = val;
        if (of16) of16[((size_t)b * TP + 1 + t) * CC + co0 + n] = (_Float16)val; // padded
    }
}

// ---------------------------------------------------------------------------
// BatchNorm (training stats)
// ---------------------------------------------------------------------------
__global__ void bn_stats(const float* __restrict__ h1, float* __restrict__ sums) {
    int c = threadIdx.x; // 256
    float s = 0.f, s2 = 0.f;
    for (int r = blockIdx.x; r < B_ * T_; r += gridDim.x) {
        float v = h1[(size_t)r * CC + c];
        s += v; s2 += v * v;
    }
    atomicAdd(&sums[c], s);
    atomicAdd(&sums[CC + c], s2);
}

__global__ void bn_final(const float* __restrict__ sums, const float* __restrict__ g,
                         const float* __restrict__ bb, float* __restrict__ scsh) {
    int c = threadIdx.x;
    float N = (float)(B_ * T_);
    float mean = sums[c] / N;
    float var  = sums[CC + c] / N - mean * mean;
    float sc   = g[c] * rsqrtf(var + EPSBN);
    scsh[c]      = sc;
    scsh[CC + c] = bb[c] - mean * sc;
}

// h1 [B][T][CC] f32 -> padded h1n [B][TP][CC] f16 with scale/shift + relu
__global__ void bn_apply(const float* __restrict__ h1, const float* __restrict__ scsh,
                         _Float16* __restrict__ h1n) {
    size_t i = (size_t)blockIdx.x * blockDim.x + threadIdx.x;
    int c = (int)(i & (CC - 1));
    float v = h1[i] * scsh[c] + scsh[CC + c];
    size_t per = (size_t)T_ * CC;
    size_t b = i / per, r = i % per;
    h1n[(b * TP + 1) * CC + r] = (_Float16)fmaxf(v, 0.f);
}

// ---------------------------------------------------------------------------
// FC: z[t][b][h] = h2[b][t][:] . fc_w[h][:] + fc_b[h], stored f16 [T][B][H]
// h2 is time-padded [B][TP][CC].
// ---------------------------------------------------------------------------
__global__ void fc_kernel(const _Float16* __restrict__ h2, const _Float16* __restrict__ fcp,
                          const float* __restrict__ fcb, _Float16* __restrict__ zf) {
    int tid  = blockIdx.x * blockDim.x + threadIdx.x;
    int lane = threadIdx.x & 31;
    int w    = tid >> 5;
    int ht   = w & 63;          // H/16 = 64
    int tt   = (w >> 6) & 63;   // T/16 = 64
    int b    = w >> 12;
    int m = lane & 15, half = lane >> 4, kb = half * 8, n = m;
    int t0 = tt * 16, h0 = ht * 16;
    const _Float16* arow = h2 + ((size_t)b * TP + 1 + t0 + m) * CC;
    v8f acc = {};
#pragma unroll
    for (int kt = 0; kt < 8; ++kt) {
        F16x16 a, bm;
        a.q[0] = *(const uint4*)(arow + kt * 32 + kb);
        a.q[1] = *(const uint4*)(arow + kt * 32 + kb + 16);
        const uint4* p = (const uint4*)fcp + (((size_t)(kt * 64 + ht) * 32 + lane) * 2);
        bm.q[0] = p[0]; bm.q[1] = p[1];
        acc = wmma_f16(a, bm, acc);
    }
    float bs = fcb[h0 + n];
#pragma unroll
    for (int v = 0; v < 8; ++v) {
        int t = t0 + v + half * 8;
        zf[((size_t)t * B_ + b) * H_ + h0 + n] = (_Float16)(acc[v] + bs);
    }
}

// ---------------------------------------------------------------------------
// Device-scope grid barrier for the persistent scan kernel.
// Fences executed by EVERY thread so each CU's L0 is flushed/invalidated.
// ---------------------------------------------------------------------------
__device__ __forceinline__ void grid_sync(unsigned* cnt, unsigned* gen, unsigned nblk) {
    __builtin_amdgcn_fence(__ATOMIC_RELEASE, "agent");   // flush my stores
    __syncthreads();                                     // whole block flushed
    if (threadIdx.x == 0) {
        unsigned g = __hip_atomic_load(gen, __ATOMIC_RELAXED, __HIP_MEMORY_SCOPE_AGENT);
        unsigned a = __hip_atomic_fetch_add(cnt, 1u, __ATOMIC_ACQ_REL, __HIP_MEMORY_SCOPE_AGENT);
        if (a == nblk - 1u) {
            __hip_atomic_store(cnt, 0u, __ATOMIC_RELAXED, __HIP_MEMORY_SCOPE_AGENT);
            __hip_atomic_fetch_add(gen, 1u, __ATOMIC_RELEASE, __HIP_MEMORY_SCOPE_AGENT);
        } else {
            while (__hip_atomic_load(gen, __ATOMIC_ACQUIRE, __HIP_MEMORY_SCOPE_AGENT) == g) {
                __builtin_amdgcn_s_sleep(2);
            }
        }
    }
    __syncthreads();
    __builtin_amdgcn_fence(__ATOMIC_ACQUIRE, "agent");   // invalidate my L0
}

// ---------------------------------------------------------------------------
// Persistent recurrent scan: 512 waves, each owns a fixed 16(b)x16(h) tile.
// Membrane/synapse state lives in VGPRs across all 1024 steps; only spikes
// (cross-wave GEMM operands) and the spike history touch memory.
// ---------------------------------------------------------------------------
__global__ void __launch_bounds__(256) scan_kernel(
        const _Float16* __restrict__ zf,
        const _Float16* __restrict__ v1p, const _Float16* __restrict__ v2p,
        const float* __restrict__ v1b, const float* __restrict__ v2b,
        _Float16* __restrict__ spk1, _Float16* __restrict__ spk2,
        unsigned char* __restrict__ hist,
        unsigned* __restrict__ barcnt, unsigned* __restrict__ bargen) {
    int lane = threadIdx.x & 31;
    int w    = (blockIdx.x * 256 + threadIdx.x) >> 5;   // 0..511
    int ht   = w & 63;
    int b0   = (w >> 6) * 16;
    int m = lane & 15, half = lane >> 4, kb = half * 8, n = m;
    int h = ht * 16 + n;
    float vb1 = v1b[h], vb2 = v2b[h];
    float m1[8], sy[8], m2[8];
#pragma unroll
    for (int v = 0; v < 8; ++v) { m1[v] = 0.f; sy[v] = 0.f; m2[v] = 0.f; }
    const size_t BH = (size_t)B_ * H_;

    for (int t = 0; t < T_; ++t) {
        const _Float16* s1in = spk1 + (size_t)(t & 1) * BH;
        const _Float16* s2in = spk2 + (size_t)(t & 1) * BH;
        _Float16* s1out = spk1 + (size_t)((t + 1) & 1) * BH;
        _Float16* s2out = spk2 + (size_t)((t + 1) & 1) * BH;

        // Hoist z_t loads: HBM latency overlaps the 64 WMMAs below.
        float zt[8];
#pragma unroll
        for (int v = 0; v < 8; ++v) {
            int b = b0 + v + half * 8;
            zt[v] = (float)zf[((size_t)t * B_ + b) * H_ + h];
        }

        const _Float16* a1row = s1in + (size_t)(b0 + m) * H_;
        const _Float16* a2row = s2in + (size_t)(b0 + m) * H_;
        v8f acc1 = {}, acc2 = {};
#pragma unroll 4
        for (int kt = 0; kt < 32; ++kt) {
            F16x16 a1, a2, b1, b2;
            a1.q[0] = *(const uint4*)(a1row + kt * 32 + kb);
            a1.q[1] = *(const uint4*)(a1row + kt * 32 + kb + 16);
            a2.q[0] = *(const uint4*)(a2row + kt * 32 + kb);
            a2.q[1] = *(const uint4*)(a2row + kt * 32 + kb + 16);
            const uint4* p1 = (const uint4*)v1p + (((size_t)(kt * 64 + ht) * 32 + lane) * 2);
            const uint4* p2 = (const uint4*)v2p + (((size_t)(kt * 64 + ht) * 32 + lane) * 2);
            b1.q[0] = p1[0]; b1.q[1] = p1[1];
            b2.q[0] = p2[0]; b2.q[1] = p2[1];
            acc1 = wmma_f16(a1, b1, acc1);
            acc2 = wmma_f16(a2, b2, acc2);
        }

#pragma unroll
        for (int v = 0; v < 8; ++v) {
            int b = b0 + v + half * 8;
            size_t idx = (size_t)b * H_ + h;
            float r1 = (m1[v] > THR) ? THR : 0.f;
            m1[v] = BETA1 * m1[v] + zt[v] + acc1[v] + vb1 - r1;
            float s1 = (m1[v] > THR) ? 1.f : 0.f;
            s1out[idx] = (_Float16)s1;
            float r2 = (m2[v] > THR) ? THR : 0.f;
            sy[v] = ALPHA2 * sy[v] + s1 + acc2[v] + vb2;
            m2[v] = BETA2 * m2[v] + sy[v] - r2;
            float s2v = (m2[v] > THR) ? 1.f : 0.f;
            s2out[idx] = (_Float16)s2v;
            hist[((size_t)t * B_ + b) * H_ + h] = (unsigned char)(m2[v] > THR ? 1 : 0);
        }

        grid_sync(barcnt, bargen, gridDim.x);
    }
}

// ---------------------------------------------------------------------------
// Head: out[b][t][k] = sum_h spk2[t][b][h] * head_w[k][h] + head_b[k]
// ---------------------------------------------------------------------------
__global__ void head_kernel(const unsigned char* __restrict__ hist,
                            const float* __restrict__ hw, const float* __restrict__ hb,
                            float* __restrict__ out) {
    int tid  = blockIdx.x * blockDim.x + threadIdx.x;
    int lane = threadIdx.x & 31;
    int w    = tid >> 5;           // w = t*B + b
    int b = w & (B_ - 1);
    int t = w / B_;
    const unsigned char* row = hist + (size_t)w * H_;
    float a0 = 0.f, a1 = 0.f, a2 = 0.f;
    for (int i = 0; i < H_ / 128; ++i) {
        unsigned u = *(const unsigned*)(row + i * 128 + lane * 4);
#pragma unroll
        for (int j = 0; j < 4; ++j) {
            float s = (float)((u >> (8 * j)) & 0xffu);
            int h = i * 128 + lane * 4 + j;
            a0 += s * hw[h];
            a1 += s * hw[H_ + h];
            a2 += s * hw[2 * H_ + h];
        }
    }
    for (int off = 16; off; off >>= 1) {
        a0 += __shfl_down(a0, off);
        a1 += __shfl_down(a1, off);
        a2 += __shfl_down(a2, off);
    }
    if (lane == 0) {
        size_t o = ((size_t)b * T_ + t) * NCLS;
        out[o]     = a0 + hb[0];
        out[o + 1] = a1 + hb[1];
        out[o + 2] = a2 + hb[2];
    }
}

// ---------------------------------------------------------------------------
extern "C" void kernel_launch(void* const* d_in, const int* in_sizes, int n_in,
                              void* d_out, int out_size, void* d_ws, size_t ws_size,
                              hipStream_t stream) {
    (void)in_sizes; (void)n_in; (void)out_size; (void)ws_size;
    const float* x   = (const float*)d_in[0];
    const float* w1  = (const float*)d_in[1];
    const float* b1  = (const float*)d_in[2];
    const float* bng = (const float*)d_in[3];
    const float* bnb = (const float*)d_in[4];
    const float* w2  = (const float*)d_in[5];
    const float* b2  = (const float*)d_in[6];
    const float* fcw = (const float*)d_in[7];
    const float* fcb = (const float*)d_in[8];
    const float* v1w = (const float*)d_in[9];
    const float* v1b = (const float*)d_in[10];
    const float* v2w = (const float*)d_in[11];
    const float* v2b = (const float*)d_in[12];
    const float* hw  = (const float*)d_in[13];
    const float* hb  = (const float*)d_in[14];
    float* out = (float*)d_out;

    char* wsb = (char*)d_ws;
    size_t off = 0;
    auto alloc = [&](size_t bytes) { size_t o = off; off = (off + bytes + 255) & ~(size_t)255; return o; };

    const size_t BT = (size_t)B_ * T_;
    _Float16* xh   = (_Float16*)(wsb + alloc((size_t)B_ * TP * CIN * 2));   // padded
    float*    h1   = (float*)   (wsb + alloc(BT * CC * 4));
    _Float16* h1n  = (_Float16*)(wsb + alloc((size_t)B_ * TP * CC * 2));    // padded
    _Float16* h2   = (_Float16*)(wsb + alloc((size_t)B_ * TP * CC * 2));    // padded
    _Float16* zf   = (_Float16*)(wsb + alloc((size_t)T_ * B_ * H_ * 2));
    unsigned char* hist = (unsigned char*)(wsb + alloc((size_t)T_ * B_ * H_));
    _Float16* spk1 = (_Float16*)(wsb + alloc((size_t)2 * B_ * H_ * 2));
    _Float16* spk2 = (_Float16*)(wsb + alloc((size_t)2 * B_ * H_ * 2));
    float*    bnsums = (float*)(wsb + alloc(2 * CC * 4));
    float*    scsh   = (float*)(wsb + alloc(2 * CC * 4));
    unsigned* barcnt = (unsigned*)(wsb + alloc(256));
    unsigned* bargen = (unsigned*)(wsb + alloc(256));
    _Float16* w1p = (_Float16*)(wsb + alloc((size_t)3 * 2 * 16 * 512 * 2));   // KT=2  NT=16 x3 d
    _Float16* w2p = (_Float16*)(wsb + alloc((size_t)3 * 8 * 16 * 512 * 2));   // KT=8  NT=16 x3 d
    _Float16* fcp = (_Float16*)(wsb + alloc((size_t)8 * 64 * 512 * 2));       // KT=8  NT=64
    _Float16* v1p = (_Float16*)(wsb + alloc((size_t)32 * 64 * 512 * 2));      // KT=32 NT=64
    _Float16* v2p = (_Float16*)(wsb + alloc((size_t)32 * 64 * 512 * 2));

    // Zero pads, spike buffers, accumulators, barrier state (capture-safe)
    hipMemsetAsync(xh,  0, (size_t)B_ * TP * CIN * 2, stream);
    hipMemsetAsync(h1n, 0, (size_t)B_ * TP * CC * 2, stream);
    hipMemsetAsync(h2,  0, (size_t)B_ * TP * CC * 2, stream);
    hipMemsetAsync(spk1, 0, (size_t)2 * B_ * H_ * 2, stream);
    hipMemsetAsync(spk2, 0, (size_t)2 * B_ * H_ * 2, stream);
    hipMemsetAsync(bnsums, 0, 2 * CC * 4, stream);
    hipMemsetAsync(barcnt, 0, 256, stream);
    hipMemsetAsync(bargen, 0, 256, stream);

    // Input f32 -> padded f16
    cvt_x_pad<<<(unsigned)(BT * CIN / 256), 256, 0, stream>>>(x, xh);

    // Pack weights into WMMA B-tile layout
    for (int d = 0; d < 3; ++d) {
        pack_b_kernel<<<(2 * 16 * 256 + 255) / 256, 256, 0, stream>>>(
            w1, w1p + (size_t)d * 2 * 16 * 512, 2, 16, CIN * 3, 3, d);
        pack_b_kernel<<<(8 * 16 * 256 + 255) / 256, 256, 0, stream>>>(
            w2, w2p + (size_t)d * 8 * 16 * 512, 8, 16, CC * 3, 3, d);
    }
    pack_b_kernel<<<(8 * 64 * 256 + 255) / 256, 256, 0, stream>>>(fcw, fcp, 8, 64, CC, 1, 0);
    pack_b_kernel<<<(32 * 64 * 256 + 255) / 256, 256, 0, stream>>>(v1w, v1p, 32, 64, H_, 1, 0);
    pack_b_kernel<<<(32 * 64 * 256 + 255) / 256, 256, 0, stream>>>(v2w, v2p, 32, 64, H_, 1, 0);

    // conv1 -> h1 (f32, pre-BN)
    conv_kernel<<<16384, 256, 0, stream>>>(xh, w1p, b1, h1, nullptr, CIN, 0);
    // BN stats + apply + relu -> h1n (padded f16)
    bn_stats<<<1024, 256, 0, stream>>>(h1, bnsums);
    bn_final<<<1, 256, 0, stream>>>(bnsums, bng, bnb, scsh);
    bn_apply<<<(unsigned)(BT * CC / 256), 256, 0, stream>>>(h1, scsh, h1n);
    // conv2 + bias + relu -> h2 (padded f16)
    conv_kernel<<<16384, 256, 0, stream>>>(h1n, w2p, b2, nullptr, h2, CC, 1);
    // fc -> zf [T][B][H] f16
    fc_kernel<<<65536, 256, 0, stream>>>(h2, fcp, fcb, zf);

    // Persistent recurrent scan: one launch, grid barrier between steps
    scan_kernel<<<NBLK, 256, 0, stream>>>(zf, v1p, v2p, v1b, v2b,
                                          spk1, spk2, hist, barcnt, bargen);

    // Head readout from spike history
    head_kernel<<<16384, 256, 0, stream>>>(hist, hw, hb, out);
}
